// MultiheadAttention_37709812858960
// MI455X (gfx1250) — compile-verified, matrix operand order checked by reading, not run on hardware
//
#include <hip/hip_runtime.h>
#include <hip/hip_bf16.h>

typedef __bf16 bf16;
typedef bf16  bf16x8  __attribute__((ext_vector_type(8)));
typedef bf16  bf16x16 __attribute__((ext_vector_type(16)));
typedef float f32x4   __attribute__((ext_vector_type(4)));
typedef float f32x8   __attribute__((ext_vector_type(8)));

#define TGT   2048
#define BSZ   4
#define EMB   1024
#define NH    16
#define HD    64
#define BH    (BSZ*NH)      // 64
#define MROWS (TGT*BSZ)     // 8192

// ---- WMMA fragment helpers (layouts per CDNA5 ISA §7.12.2) ----
// A 16x32 bf16: lane holds row M=lane&15; K chunks [hi*8, hi*8+8) and [16+hi*8, ...).
// B 32x16 bf16 (dual): lane holds col N=lane&15, same K chunking.
// C/D 16x16 f32: element v <-> row hi*8+v, col lane&15.

__device__ __forceinline__ bf16x16 frag_ld_bf16(const bf16* rp, int hi) {
    const bf16x8* p0 = (const bf16x8*)(rp + hi * 8);
    const bf16x8* p1 = (const bf16x8*)(rp + hi * 8 + 16);
    bf16x8 a = *p0, b = *p1;
    bf16x16 r;
#pragma unroll
    for (int i = 0; i < 8; ++i) { r[i] = a[i]; r[i + 8] = b[i]; }
    return r;
}

__device__ __forceinline__ bf16x16 frag_ld_f32cvt(const float* rp, int hi) {
    const f32x4* p0 = (const f32x4*)(rp + hi * 8);
    const f32x4* p1 = (const f32x4*)(rp + hi * 8 + 16);
    f32x4 x0 = p0[0], x1 = p0[1], x2 = p1[0], x3 = p1[1];
    bf16x16 r;
#pragma unroll
    for (int i = 0; i < 4; ++i) {
        r[i]      = (bf16)x0[i];
        r[i + 4]  = (bf16)x1[i];
        r[i + 8]  = (bf16)x2[i];
        r[i + 12] = (bf16)x3[i];
    }
    return r;
}

__device__ __forceinline__ f32x8 wmma_bf16(bf16x16 a, bf16x16 b, f32x8 c) {
    return __builtin_amdgcn_wmma_f32_16x16x32_bf16(false, a, false, b, (short)0, c,
                                                   false, false);
}

// ---- CDNA5 async global->LDS copy (ASYNCcnt path, CDNA5 ISA) ----
// 16 bytes per lane. LDS offset = low 32 bits of the generic pointer
// (aperture base occupies addr[63:32]).
__device__ __forceinline__ void async_copy16(const bf16* gsrc, bf16* ldst) {
    unsigned loff = (unsigned)(unsigned long long)ldst;
    asm volatile("global_load_async_to_lds_b128 %0, %1, off"
                 :: "v"(loff), "v"(gsrc)
                 : "memory");
}
__device__ __forceinline__ void wait_async0() {
    asm volatile("s_wait_asynccnt 0" ::: "memory");
}

// ---------------- K0: weight f32 -> bf16 ----------------
__global__ void mha_cvt_bf16(const float* __restrict__ src, bf16* __restrict__ dst, int n) {
    int i = blockIdx.x * blockDim.x + threadIdx.x;
    if (i < n) dst[i] = (bf16)src[i];
}

// ---------------- K1: QKV projection (wave tile 32Mx64N, B reused 2x) ----------------
__global__ void __launch_bounds__(256)
mha_qkv_proj(const float* __restrict__ q_in, const float* __restrict__ k_in,
             const float* __restrict__ v_in, const bf16* __restrict__ Wbf,
             const float* __restrict__ bias,
             bf16* __restrict__ Qbf, bf16* __restrict__ Kbf, bf16* __restrict__ Vt) {
    const int lane = threadIdx.x & 31;
    const int wv   = threadIdx.x >> 5;
    const int r    = lane & 15;
    const int hi   = lane >> 4;
    const int sec  = blockIdx.z;
    const int n0   = blockIdx.y * 64;
    const int m0   = blockIdx.x * 256 + wv * 32;

    const float* X = (sec == 0) ? q_in : (sec == 1) ? k_in : v_in;
    const bf16*  W = Wbf + (size_t)(sec * 1024 + n0) * EMB;

    f32x8 acc[2][4] = {};
    for (int k = 0; k < EMB; k += 32) {
        bf16x16 a0 = frag_ld_f32cvt(X + (size_t)(m0 + r) * EMB + k, hi);
        bf16x16 a1 = frag_ld_f32cvt(X + (size_t)(m0 + 16 + r) * EMB + k, hi);
#pragma unroll
        for (int nt = 0; nt < 4; ++nt) {
            bf16x16 bw = frag_ld_bf16(W + (size_t)(nt * 16 + r) * EMB + k, hi);
            acc[0][nt] = wmma_bf16(a0, bw, acc[0][nt]);
            acc[1][nt] = wmma_bf16(a1, bw, acc[1][nt]);
        }
    }

    // Section branch hoisted OUT of the store loops -> three straight-line epilogues.
    if (sec == 0) {
        const float qscale = 0.125f;  // 1/sqrt(HD)
#pragma unroll
        for (int mt = 0; mt < 2; ++mt)
#pragma unroll
            for (int nt = 0; nt < 4; ++nt) {
                const int n  = n0 + nt * 16 + r;
                const float bs = bias[n];
                const int h = n >> 6, d = n & 63;
#pragma unroll
                for (int v = 0; v < 8; ++v) {
                    const int m = m0 + mt * 16 + hi * 8 + v;
                    const int t = m >> 2, b = m & 3;
                    Qbf[((size_t)(b * NH + h) * TGT + t) * HD + d] =
                        (bf16)((acc[mt][nt][v] + bs) * qscale);
                }
            }
    } else if (sec == 1) {
#pragma unroll
        for (int mt = 0; mt < 2; ++mt)
#pragma unroll
            for (int nt = 0; nt < 4; ++nt) {
                const int n  = n0 + nt * 16 + r;
                const float bs = bias[1024 + n];
                const int h = n >> 6, d = n & 63;
#pragma unroll
                for (int v = 0; v < 8; ++v) {
                    const int m = m0 + mt * 16 + hi * 8 + v;
                    const int t = m >> 2, b = m & 3;
                    Kbf[((size_t)(b * NH + h) * TGT + t) * HD + d] =
                        (bf16)(acc[mt][nt][v] + bs);
                }
            }
    } else {
#pragma unroll
        for (int mt = 0; mt < 2; ++mt)
#pragma unroll
            for (int nt = 0; nt < 4; ++nt) {
                const int n  = n0 + nt * 16 + r;
                const float bs = bias[2048 + n];
                const int h = n >> 6, d = n & 63;
#pragma unroll
                for (int v = 0; v < 8; ++v) {
                    const int m = m0 + mt * 16 + hi * 8 + v;
                    const int t = m >> 2, b = m & 3;
                    Vt[((size_t)(b * NH + h) * HD + d) * TGT + t] =
                        (bf16)(acc[mt][nt][v] + bs);   // transposed V
                }
            }
    }
}

// ---------------- K2: softmax stats (K tiles async-staged to LDS) ----------------
__global__ void __launch_bounds__(256)
mha_attn_stats(const bf16* __restrict__ Qbf, const bf16* __restrict__ Kbf,
               float* __restrict__ Mbuf, float* __restrict__ Lbuf) {
    __shared__ __align__(16) bf16 kbuf[2][32][HD];      // 2 x 4 KB
    const int lane = threadIdx.x & 31, wv = threadIdx.x >> 5;
    const int r = lane & 15, hi = lane >> 4;
    const int bh = blockIdx.x;
    const int i0 = (blockIdx.y * 8 + wv) * 16;
    const bf16* Q = Qbf + ((size_t)bh * TGT + i0) * HD;
    const bf16* K = Kbf + (size_t)bh * TGT * HD;

    const int crow = threadIdx.x >> 3, cch = (threadIdx.x & 7) * 8;  // 256 lanes -> 32x64 tile

    bf16x16 a0 = frag_ld_bf16(Q + (size_t)r * HD, hi);
    bf16x16 a1 = frag_ld_bf16(Q + (size_t)r * HD + 32, hi);

    float m8[8], l8[8];
#pragma unroll
    for (int v = 0; v < 8; ++v) { m8[v] = -3.0e38f; l8[v] = 0.f; }

    async_copy16(K + (size_t)crow * HD + cch, &kbuf[0][crow][cch]);  // prologue stage 0

    for (int j0 = 0; j0 < TGT; j0 += 32) {
        const int cur = (j0 >> 5) & 1;
        wait_async0();
        __syncthreads();
        if (j0 + 32 < TGT)
            async_copy16(K + (size_t)(j0 + 32 + crow) * HD + cch, &kbuf[cur ^ 1][crow][cch]);

        f32x8 s0 = {}, s1 = {};
        bf16x16 b00 = frag_ld_bf16(&kbuf[cur][r][0], hi);
        bf16x16 b01 = frag_ld_bf16(&kbuf[cur][r][32], hi);
        s0 = wmma_bf16(a0, b00, s0);
        s0 = wmma_bf16(a1, b01, s0);
        bf16x16 b10 = frag_ld_bf16(&kbuf[cur][16 + r][0], hi);
        bf16x16 b11 = frag_ld_bf16(&kbuf[cur][16 + r][32], hi);
        s1 = wmma_bf16(a0, b10, s1);
        s1 = wmma_bf16(a1, b11, s1);

#pragma unroll
        for (int v = 0; v < 8; ++v) {
            float cm = fmaxf(s0[v], s1[v]);
#pragma unroll
            for (int mk = 8; mk >= 1; mk >>= 1) cm = fmaxf(cm, __shfl_xor(cm, mk, 32));
            const float mnew = fmaxf(m8[v], cm);
            float e = __expf(s0[v] - mnew) + __expf(s1[v] - mnew);
#pragma unroll
            for (int mk = 8; mk >= 1; mk >>= 1) e += __shfl_xor(e, mk, 32);
            l8[v] = l8[v] * __expf(m8[v] - mnew) + e;
            m8[v] = mnew;
        }
    }
    if (r == 0) {
#pragma unroll
        for (int v = 0; v < 8; ++v) {
            Mbuf[(size_t)bh * TGT + i0 + hi * 8 + v] = m8[v];
            Lbuf[(size_t)bh * TGT + i0 + hi * 8 + v] = l8[v];
        }
    }
}

// ---------------- K3: O = softmax(S) @ V  (K/V async-staged, P via LDS transpose) ----
__global__ void __launch_bounds__(256)
mha_attn_out(const bf16* __restrict__ Qbf, const bf16* __restrict__ Kbf,
             const bf16* __restrict__ Vt, const float* __restrict__ Mbuf,
             const float* __restrict__ Lbuf, bf16* __restrict__ Ctx) {
    __shared__ __align__(16) bf16 kbuf[2][32][HD];      // 2 x 4 KB
    __shared__ __align__(16) bf16 vbuf[2][HD][32];      // 2 x 4 KB (Vt tile: rows=d)
    __shared__ __align__(16) bf16 plds[8][16][32];      // 8 KB (per-wave P transpose)
    const int lane = threadIdx.x & 31, wv = threadIdx.x >> 5;
    const int r = lane & 15, hi = lane >> 4;
    const int bh = blockIdx.x;
    const int i0 = (blockIdx.y * 8 + wv) * 16;
    const bf16* Q = Qbf + ((size_t)bh * TGT + i0) * HD;
    const bf16* K = Kbf + (size_t)bh * TGT * HD;
    const bf16* V = Vt + (size_t)bh * HD * TGT;

    const int krow = threadIdx.x >> 3, kch = (threadIdx.x & 7) * 8;  // 32x64 K tile
    const int vrow = threadIdx.x >> 2, vch = (threadIdx.x & 3) * 8;  // 64x32 V tile

    bf16x16 a0 = frag_ld_bf16(Q + (size_t)r * HD, hi);
    bf16x16 a1 = frag_ld_bf16(Q + (size_t)r * HD + 32, hi);

    float mrow[8], linv[8];
#pragma unroll
    for (int v = 0; v < 8; ++v) {
        mrow[v] = Mbuf[(size_t)bh * TGT + i0 + hi * 8 + v];
        linv[v] = 1.0f / Lbuf[(size_t)bh * TGT + i0 + hi * 8 + v];
    }

    // prologue stage 0
    async_copy16(K + (size_t)krow * HD + kch, &kbuf[0][krow][kch]);
    async_copy16(V + (size_t)vrow * TGT + vch, &vbuf[0][vrow][vch]);

    f32x8 o[4] = {};
    for (int j0 = 0; j0 < TGT; j0 += 32) {
        const int cur = (j0 >> 5) & 1;
        wait_async0();
        __syncthreads();
        if (j0 + 32 < TGT) {
            async_copy16(K + (size_t)(j0 + 32 + krow) * HD + kch, &kbuf[cur ^ 1][krow][kch]);
            async_copy16(V + (size_t)vrow * TGT + j0 + 32 + vch, &vbuf[cur ^ 1][vrow][vch]);
        }

        f32x8 s0 = {}, s1 = {};
        bf16x16 b00 = frag_ld_bf16(&kbuf[cur][r][0], hi);
        bf16x16 b01 = frag_ld_bf16(&kbuf[cur][r][32], hi);
        s0 = wmma_bf16(a0, b00, s0);
        s0 = wmma_bf16(a1, b01, s0);
        bf16x16 b10 = frag_ld_bf16(&kbuf[cur][16 + r][0], hi);
        bf16x16 b11 = frag_ld_bf16(&kbuf[cur][16 + r][32], hi);
        s1 = wmma_bf16(a0, b10, s1);
        s1 = wmma_bf16(a1, b11, s1);

        // C-fragment -> A-fragment transpose of the 16x32 P tile through LDS
#pragma unroll
        for (int v = 0; v < 8; ++v) {
            plds[wv][hi * 8 + v][r]      = (bf16)(__expf(s0[v] - mrow[v]) * linv[v]);
            plds[wv][hi * 8 + v][16 + r] = (bf16)(__expf(s1[v] - mrow[v]) * linv[v]);
        }
        __builtin_amdgcn_wave_barrier();   // per-wave region; DS ops are in-order
        bf16x16 ap = frag_ld_bf16(&plds[wv][r][0], hi);
        __builtin_amdgcn_wave_barrier();

#pragma unroll
        for (int dt = 0; dt < 4; ++dt) {
            bf16x16 bv = frag_ld_bf16(&vbuf[cur][dt * 16 + r][0], hi);
            o[dt] = wmma_bf16(ap, bv, o[dt]);
        }
    }

    const int b = bh >> 4, h = bh & 15;
#pragma unroll
    for (int dt = 0; dt < 4; ++dt) {
        const int e = h * HD + dt * 16 + r;
#pragma unroll
        for (int v = 0; v < 8; ++v) {
            const int t = i0 + hi * 8 + v;
            Ctx[((size_t)t * BSZ + b) * EMB + e] = (bf16)o[dt][v];
        }
    }
}

// ---------------- K4: avg_weights = mean over heads of softmax(S) ----------------
__global__ void __launch_bounds__(256)
mha_attn_avg(const bf16* __restrict__ Qbf, const bf16* __restrict__ Kbf,
             const float* __restrict__ Mbuf, const float* __restrict__ Lbuf,
             float* __restrict__ avg) {
    const int lane = threadIdx.x & 31, wv = threadIdx.x >> 5;
    const int r = lane & 15, hi = lane >> 4;
    const int b  = blockIdx.x;                       // batch
    const int i0 = (blockIdx.y * 8 + wv) * 16;

    for (int j0 = 0; j0 < TGT; j0 += 32) {
        float sum0[8] = {}, sum1[8] = {};
        for (int h = 0; h < NH; ++h) {
            const int bh = b * NH + h;
            const bf16* Q = Qbf + ((size_t)bh * TGT + i0) * HD;
            const bf16* K = Kbf + (size_t)bh * TGT * HD;
            // prefetch next head's K tile (global_prefetch_b8)
            __builtin_prefetch((const void*)(K + (size_t)(TGT + j0 + r) * HD), 0, 1);
            bf16x16 a0 = frag_ld_bf16(Q + (size_t)r * HD, hi);
            bf16x16 a1 = frag_ld_bf16(Q + (size_t)r * HD + 32, hi);

            f32x8 s0 = {}, s1 = {};
            bf16x16 b00 = frag_ld_bf16(K + (size_t)(j0 + r) * HD, hi);
            bf16x16 b01 = frag_ld_bf16(K + (size_t)(j0 + r) * HD + 32, hi);
            s0 = wmma_bf16(a0, b00, s0);
            s0 = wmma_bf16(a1, b01, s0);
            bf16x16 b10 = frag_ld_bf16(K + (size_t)(j0 + 16 + r) * HD, hi);
            bf16x16 b11 = frag_ld_bf16(K + (size_t)(j0 + 16 + r) * HD + 32, hi);
            s1 = wmma_bf16(a0, b10, s1);
            s1 = wmma_bf16(a1, b11, s1);

#pragma unroll
            for (int v = 0; v < 8; ++v) {
                const float mr = Mbuf[(size_t)bh * TGT + i0 + hi * 8 + v];
                const float li = 1.0f / Lbuf[(size_t)bh * TGT + i0 + hi * 8 + v];
                sum0[v] += __expf(s0[v] - mr) * li;
                sum1[v] += __expf(s1[v] - mr) * li;
            }
        }
        const float ih = 1.0f / (float)NH;
#pragma unroll
        for (int v = 0; v < 8; ++v) {
            const int t = i0 + hi * 8 + v;
            const size_t base = ((size_t)b * TGT + t) * TGT + j0;
            avg[base + r]      = sum0[v] * ih;
            avg[base + 16 + r] = sum1[v] * ih;
        }
    }
}

// ---------------- K5: output projection (wave tile 32Mx64N) ----------------
__global__ void __launch_bounds__(256)
mha_out_proj(const bf16* __restrict__ Ctx, const bf16* __restrict__ OW,
             const float* __restrict__ bias, float* __restrict__ out) {
    const int lane = threadIdx.x & 31, wv = threadIdx.x >> 5;
    const int r = lane & 15, hi = lane >> 4;
    const int n0 = blockIdx.y * 64;
    const int m0 = blockIdx.x * 256 + wv * 32;

    f32x8 acc[2][4] = {};
    for (int k = 0; k < EMB; k += 32) {
        bf16x16 a0 = frag_ld_bf16(Ctx + (size_t)(m0 + r) * EMB + k, hi);
        bf16x16 a1 = frag_ld_bf16(Ctx + (size_t)(m0 + 16 + r) * EMB + k, hi);
#pragma unroll
        for (int nt = 0; nt < 4; ++nt) {
            bf16x16 bw = frag_ld_bf16(OW + (size_t)(n0 + nt * 16 + r) * EMB + k, hi);
            acc[0][nt] = wmma_bf16(a0, bw, acc[0][nt]);
            acc[1][nt] = wmma_bf16(a1, bw, acc[1][nt]);
        }
    }
#pragma unroll
    for (int mt = 0; mt < 2; ++mt) {
#pragma unroll
        for (int nt = 0; nt < 4; ++nt) {
            const int n = n0 + nt * 16 + r;
            const float bs = bias[n];
#pragma unroll
            for (int v = 0; v < 8; ++v) {
                const int m = m0 + mt * 16 + hi * 8 + v;
                out[(size_t)m * EMB + n] = acc[mt][nt][v] + bs;
            }
        }
    }
}

// ---------------- host launch ----------------
extern "C" void kernel_launch(void* const* d_in, const int* in_sizes, int n_in,
                              void* d_out, int out_size, void* d_ws, size_t ws_size,
                              hipStream_t stream) {
    (void)in_sizes; (void)n_in; (void)out_size; (void)ws_size;

    const float* q_in  = (const float*)d_in[0];
    const float* k_in  = (const float*)d_in[1];
    const float* v_in  = (const float*)d_in[2];
    const float* wIn   = (const float*)d_in[3];   // (3072, 1024)
    const float* bIn   = (const float*)d_in[4];   // (3072,)
    const float* wOut  = (const float*)d_in[5];   // (1024, 1024)
    const float* bOut  = (const float*)d_in[6];   // (1024,)

    float* out = (float*)d_out;                            // (2048,4,1024)
    float* avg = (float*)d_out + (size_t)MROWS * EMB;      // (4,2048,2048)

    char* ws = (char*)d_ws;
    size_t off = 0;
    bf16*  Qbf  = (bf16*)(ws + off);  off += (size_t)BH * TGT * HD * sizeof(bf16);   // 16 MB
    bf16*  Kbf  = (bf16*)(ws + off);  off += (size_t)BH * TGT * HD * sizeof(bf16);   // 16 MB
    bf16*  Vt   = (bf16*)(ws + off);  off += (size_t)BH * HD * TGT * sizeof(bf16);   // 16 MB
    bf16*  Ctx  = (bf16*)(ws + off);  off += (size_t)MROWS * EMB * sizeof(bf16);     // 16 MB
    bf16*  Wbf  = (bf16*)(ws + off);  off += (size_t)3 * EMB * EMB * sizeof(bf16);   //  6 MB
    bf16*  OWbf = (bf16*)(ws + off);  off += (size_t)EMB * EMB * sizeof(bf16);       //  2 MB
    float* Mbuf = (float*)(ws + off); off += (size_t)BH * TGT * sizeof(float);
    float* Lbuf = (float*)(ws + off); off += (size_t)BH * TGT * sizeof(float);

    // K0: weight conversion
    {
        const int n1 = 3 * EMB * EMB;
        const int n2 = EMB * EMB;
        mha_cvt_bf16<<<(n1 + 255) / 256, 256, 0, stream>>>(wIn, Wbf, n1);
        mha_cvt_bf16<<<(n2 + 255) / 256, 256, 0, stream>>>(wOut, OWbf, n2);
    }
    // K1: QKV projection — grid (M/256, 1024/64, 3 sections)
    mha_qkv_proj<<<dim3(MROWS / 256, 1024 / 64, 3), 256, 0, stream>>>(
        q_in, k_in, v_in, Wbf, bIn, Qbf, Kbf, Vt);
    // K2: softmax stats — grid (BH, TGT/(16*8))
    mha_attn_stats<<<dim3(BH, TGT / 128), 256, 0, stream>>>(Qbf, Kbf, Mbuf, Lbuf);
    // K3: attention output — grid (BH, TGT/128)
    mha_attn_out<<<dim3(BH, TGT / 128), 256, 0, stream>>>(Qbf, Kbf, Vt, Mbuf, Lbuf, Ctx);
    // K4: averaged attention weights — grid (BSZ, TGT/128)
    mha_attn_avg<<<dim3(BSZ, TGT / 128), 256, 0, stream>>>(Qbf, Kbf, Mbuf, Lbuf, avg);
    // K5: output projection — grid (M/256, 1024/64)
    mha_out_proj<<<dim3(MROWS / 256, EMB / 64), 256, 0, stream>>>(Ctx, OWbf, bOut, out);
}
